// SQALSTM_81595788689546
// MI455X (gfx1250) — matrix operand order repaired and checked by reading, not compiled, not run on hardware
//
#include <hip/hip_runtime.h>
#include <hip/hip_bf16.h>

// ---------------------------------------------------------------------------
// SQALSTM for MI455X (gfx1250, wave32, WMMA).
//   Layer pipeline (per layer, all on `stream`):
//     1) qa  = q0*Wqh + a0*Wah + bias_sum          (WMMA bf16, M=64)
//     2) xg  = X*Wih  + qa                         (WMMA bf16, M=32768)
//     3) persistent recurrent kernel: 512 steps of gates = xg[t] + h*Whh,
//        LSTM cell elementwise, grid barrier between steps.
//   All GEMM operands converted to bf16 (fp32 accumulate). B operands are
//   pre-transposed to [N,K] so WMMA B fragments are contiguous per lane.
// ---------------------------------------------------------------------------

typedef __attribute__((ext_vector_type(16))) __bf16 v16bf;
typedef __attribute__((ext_vector_type(8)))  float  v8f;

#define T_STEPS 512
#define BATCH   64
#define HID     1024
#define G4      4096
#define LAYERS  2
#define NB_REC  32      // persistent blocks in recurrent kernel (must be co-resident)

static __device__ __forceinline__ __bf16 f2bf(float x) {
  union { float f; unsigned u; } v; v.f = x;
  unsigned r = v.u + 0x7FFFu + ((v.u >> 16) & 1u);   // round-to-nearest-even
  union { unsigned short s; __bf16 b; } o;
  o.s = (unsigned short)(r >> 16);
  return o.b;
}
static __device__ __forceinline__ float sigm(float x) {
  return 1.0f / (1.0f + __expf(-x));
}

// A fragment: 16x32 bf16, row-major source with leading dim `ld` (elements).
// ISA layout: lanes 0-15 row=lane, K chunks [0..7] and [16..23];
//             lanes 16-31 row=lane-16, K chunks [8..15] and [24..31].
static __device__ __forceinline__ v16bf load_frag_a(const __bf16* base, int ld) {
  const int lane = threadIdx.x & 31;
  const int row  = lane & 15;
  const int k0   = (lane >> 4) ? 8 : 0;
  const char* p = (const char*)(base + row * ld + k0);
  union { v16bf v; uint4 u[2]; } f;
  f.u[0] = *(const uint4*)(p);
  f.u[1] = *(const uint4*)(p + 32);   // +16 elements
  return f.v;
}

// B fragment: 32x16 bf16 from a transposed ([N,K] row-major) source.
// ISA layout: lanes 0-15 col=lane K [0..7],[16..23]; lanes 16-31 col=lane-16 K [8..15],[24..31].
static __device__ __forceinline__ v16bf load_frag_b(const __bf16* baseT, int ld) {
  const int lane = threadIdx.x & 31;
  const int col  = lane & 15;
  const int k0   = (lane >> 4) ? 8 : 0;
  const char* p = (const char*)(baseT + col * ld + k0);
  union { v16bf v; uint4 u[2]; } f;
  f.u[0] = *(const uint4*)(p);
  f.u[1] = *(const uint4*)(p + 32);   // +16 elements
  return f.v;
}

// ---------------------------------------------------------------------------
__global__ void k_convert_bf16(const float* __restrict__ src,
                               __bf16* __restrict__ dst, long n) {
  long i = (long)blockIdx.x * blockDim.x + threadIdx.x;
  const long stride = (long)gridDim.x * blockDim.x;
  for (; i < n; i += stride) dst[i] = f2bf(src[i]);
}

// src [K,N] f32 row-major -> dst [N,K] bf16 row-major (LDS tiled)
__global__ void k_transpose_bf16(const float* __restrict__ src,
                                 __bf16* __restrict__ dst, int K, int N) {
  __shared__ __bf16 tile[64][72];
  const int k0 = blockIdx.y * 64, n0 = blockIdx.x * 64;
  for (int idx = threadIdx.x; idx < 64 * 64; idx += blockDim.x) {
    int r = idx >> 6, c = idx & 63;
    tile[r][c] = f2bf(src[(long)(k0 + r) * N + (n0 + c)]);
  }
  __syncthreads();
  for (int idx = threadIdx.x; idx < 64 * 64; idx += blockDim.x) {
    int r = idx >> 6, c = idx & 63;
    dst[(long)(n0 + r) * K + (k0 + c)] = tile[c][r];
  }
}

__global__ void k_bias_sum(const float* __restrict__ a, const float* __restrict__ b,
                           const float* __restrict__ c, const float* __restrict__ d,
                           float* __restrict__ out, int n) {
  int i = blockIdx.x * blockDim.x + threadIdx.x;
  if (i < n) out[i] = a[i] + b[i] + c[i] + d[i];
}

// qa[64,4096] = q0*Wqh + a0*Wah + bias_sum.  grid.x = 32 (128 cols each), 8 waves.
__global__ __launch_bounds__(256) void k_gemm_qa(
    const __bf16* __restrict__ q0, const __bf16* __restrict__ a0,
    const __bf16* __restrict__ WqhT, const __bf16* __restrict__ WahT,
    const float* __restrict__ bias_sum, float* __restrict__ qa) {
  extern __shared__ char smem_raw[];
  __bf16* astage = (__bf16*)smem_raw;                   // 64 x 1024 bf16 (128KB)
  const int tid = threadIdx.x, wave = tid >> 5, lane = tid & 31;
  const int col0 = blockIdx.x * 128 + wave * 16;
  const int n = lane & 15, m0 = (lane >> 4) ? 8 : 0;

  v8f acc[4];
#pragma unroll
  for (int m = 0; m < 4; ++m)
#pragma unroll
    for (int r = 0; r < 8; ++r) acc[m][r] = 0.0f;

  const __bf16* Asrc[2] = { q0, a0 };
  const __bf16* Bsrc[2] = { WqhT, WahT };
  for (int p = 0; p < 2; ++p) {
    __syncthreads();
    const uint4* s = (const uint4*)Asrc[p];
    uint4* dsh = (uint4*)astage;
    for (int i = tid; i < 8192; i += 256) dsh[i] = s[i];
    __syncthreads();
    const __bf16* Bt = Bsrc[p] + (size_t)col0 * HID;
    for (int kk = 0; kk < 32; ++kk) {
      __builtin_prefetch(Bt + (size_t)n * HID + (kk + 1) * 32, 0, 1);
      v16bf b = load_frag_b(Bt + kk * 32, HID);
#pragma unroll
      for (int m = 0; m < 4; ++m) {
        v16bf a = load_frag_a(astage + m * 16 * HID + kk * 32, HID);
        acc[m] = __builtin_amdgcn_wmma_f32_16x16x32_bf16(
            false, a, false, b, (short)0, acc[m], false, false);
      }
    }
  }
  const float bs = bias_sum[col0 + n];
#pragma unroll
  for (int m = 0; m < 4; ++m)
#pragma unroll
    for (int r = 0; r < 8; ++r)
      qa[(size_t)(m * 16 + m0 + r) * G4 + col0 + n] = acc[m][r] + bs;
}

// xg[32768,4096] = act*Wih + qa.  grid = (32, 512), block tile 64x128, 8 waves.
__global__ __launch_bounds__(256) void k_gemm_xg(
    const __bf16* __restrict__ act,       // [T*B, 1024] bf16
    const __bf16* __restrict__ WihT,      // [4096, 1024] bf16
    const float* __restrict__ qa,         // [64, 4096]
    float* __restrict__ xg) {             // [T*B, 4096]
  extern __shared__ char smem_raw[];
  __bf16* astage = (__bf16*)smem_raw;                   // 64 x 1024 bf16 (128KB)
  const int tid = threadIdx.x, wave = tid >> 5, lane = tid & 31;
  const int col0 = blockIdx.x * 128 + wave * 16;
  const size_t row0 = (size_t)blockIdx.y * 64;
  const int n = lane & 15, m0 = (lane >> 4) ? 8 : 0;

  const uint4* s = (const uint4*)(act + row0 * HID);
  uint4* dsh = (uint4*)astage;
  for (int i = tid; i < 8192; i += 256) dsh[i] = s[i];
  __syncthreads();

  v8f acc[4];   // init from qa (row0 is a multiple of 64 -> batch index == m)
#pragma unroll
  for (int m = 0; m < 4; ++m)
#pragma unroll
    for (int r = 0; r < 8; ++r)
      acc[m][r] = qa[(size_t)(m * 16 + m0 + r) * G4 + col0 + n];

  const __bf16* Bt = WihT + (size_t)col0 * HID;
  for (int kk = 0; kk < 32; ++kk) {
    __builtin_prefetch(Bt + (size_t)n * HID + (kk + 1) * 32, 0, 1);
    v16bf b = load_frag_b(Bt + kk * 32, HID);
#pragma unroll
    for (int m = 0; m < 4; ++m) {
      v16bf a = load_frag_a(astage + m * 16 * HID + kk * 32, HID);
      acc[m] = __builtin_amdgcn_wmma_f32_16x16x32_bf16(
          false, a, false, b, (short)0, acc[m], false, false);
    }
  }
#pragma unroll
  for (int m = 0; m < 4; ++m)
#pragma unroll
    for (int r = 0; r < 8; ++r)
      xg[(row0 + m * 16 + m0 + r) * (size_t)G4 + col0 + n] = acc[m][r];
}

__global__ void k_init_state(const float* __restrict__ h0l, const float* __restrict__ c0l,
                             __bf16* __restrict__ h_bf, float* __restrict__ c_state,
                             unsigned* __restrict__ bar) {
  const int i = blockIdx.x * blockDim.x + threadIdx.x;
  if (i < BATCH * HID) { h_bf[i] = f2bf(h0l[i]); c_state[i] = c0l[i]; }
  if (i == 0) { bar[0] = 0u; bar[1] = 0u; }
}

// Persistent recurrent kernel: NB_REC blocks; block b owns h-cols [b*32,b*32+32)
// and the 4 matching gate column groups.  LDS: full h_t stage (128KB, huge
// 320KB WGP LDS makes this cheap) + fp32 gate exchange tile (32KB).
__global__ __launch_bounds__(256) void k_recurrent(
    const float* __restrict__ xg,        // [T, B, 4096]
    const __bf16* __restrict__ WhhT,     // [4096, 1024] bf16
    __bf16* __restrict__ h_bf,           // [2, B, 1024] ping-pong
    float* __restrict__ c_state,         // [B, 1024]
    __bf16* __restrict__ hs_bf,          // [T, B, 1024] or nullptr
    float* __restrict__ hs_f32,          // [T, B, 1024] or nullptr
    float* __restrict__ h_last,          // [B, 1024]
    float* __restrict__ c_last,          // [B, 1024]
    unsigned* __restrict__ bar) {
  extern __shared__ char smem_raw[];
  __bf16* hstage = (__bf16*)smem_raw;                     // 64 x 1024 bf16
  float*  gstage = (float*)(smem_raw + 64 * 1024 * 2);    // 64 x 128  f32
  const int tid = threadIdx.x, wave = tid >> 5, lane = tid & 31;
  const int blk = blockIdx.x;
  const int g = wave >> 1, nt = wave & 1;
  const int gcol0 = g * HID + blk * 32 + nt * 16;         // this wave's gate cols
  const __bf16* Bt = WhhT + (size_t)gcol0 * HID;
  const int n = lane & 15, m0 = (lane >> 4) ? 8 : 0;

  for (int t = 0; t < T_STEPS; ++t) {
    // stage h_t into LDS (cooperative 128KB copy)
    const uint4* hsrc = (const uint4*)(h_bf + (size_t)(t & 1) * BATCH * HID);
    uint4* dsh = (uint4*)hstage;
    for (int i = tid; i < 8192; i += 256) dsh[i] = hsrc[i];
    __syncthreads();

    // gates tile = xg[t] tile + h_t * Whh slice (bf16 WMMA, f32 accumulate)
    const float* xgt = xg + (size_t)t * BATCH * G4;
    v8f acc[4];
#pragma unroll
    for (int m = 0; m < 4; ++m)
#pragma unroll
      for (int r = 0; r < 8; ++r)
        acc[m][r] = xgt[(size_t)(m * 16 + m0 + r) * G4 + gcol0 + n];

    for (int kk = 0; kk < 32; ++kk) {
      __builtin_prefetch(Bt + (size_t)n * HID + (kk + 1) * 32, 0, 1);
      v16bf b = load_frag_b(Bt + kk * 32, HID);
#pragma unroll
      for (int m = 0; m < 4; ++m) {
        v16bf a = load_frag_a(hstage + m * 16 * HID + kk * 32, HID);
        acc[m] = __builtin_amdgcn_wmma_f32_16x16x32_bf16(
            false, a, false, b, (short)0, acc[m], false, false);
      }
    }

    // exchange gates through LDS: local col = g*32 + (nt*16 + n), hc in [0,32)
#pragma unroll
    for (int m = 0; m < 4; ++m)
#pragma unroll
      for (int r = 0; r < 8; ++r)
        gstage[(m * 16 + m0 + r) * 128 + g * 32 + nt * 16 + n] = acc[m][r];
    __syncthreads();

    // LSTM cell elementwise on this block's 64 rows x 32 h-cols
#pragma unroll
    for (int e = 0; e < 8; ++e) {
      const int idx = tid + e * 256;
      const int row = idx >> 5, hc = idx & 31;
      const int hcol = blk * 32 + hc;
      const float ig = gstage[row * 128 + hc];
      const float fg = gstage[row * 128 + 32 + hc];
      const float cc = gstage[row * 128 + 64 + hc];
      const float og = gstage[row * 128 + 96 + hc];
      const float cold = c_state[row * HID + hcol];
      const float cnew = sigm(fg) * cold + sigm(ig) * tanhf(cc);
      const float hnew = sigm(og) * tanhf(cnew);
      c_state[row * HID + hcol] = cnew;
      h_bf[(size_t)((t + 1) & 1) * BATCH * HID + row * HID + hcol] = f2bf(hnew);
      if (hs_bf)  hs_bf[(size_t)t * BATCH * HID + row * HID + hcol] = f2bf(hnew);
      if (hs_f32) hs_f32[(size_t)t * BATCH * HID + row * HID + hcol] = hnew;
      if (t == T_STEPS - 1) {
        h_last[row * HID + hcol] = hnew;
        c_last[row * HID + hcol] = cnew;
      }
    }

    // grid-wide barrier between timesteps (generation-counted)
    __syncthreads();
    if (tid == 0) {
      __threadfence();
      const unsigned gen = atomicAdd(&bar[1], 0u);
      if (atomicAdd(&bar[0], 1u) == NB_REC - 1u) {
        atomicExch(&bar[0], 0u);
        atomicAdd(&bar[1], 1u);
      } else {
        while (atomicAdd(&bar[1], 0u) == gen) __builtin_amdgcn_s_sleep(1);
      }
      __threadfence();
    }
    __syncthreads();
  }
}

// ---------------------------------------------------------------------------
extern "C" void kernel_launch(void* const* d_in, const int* in_sizes, int n_in,
                              void* d_out, int out_size, void* d_ws, size_t ws_size,
                              hipStream_t stream) {
  (void)in_sizes; (void)n_in; (void)out_size; (void)ws_size;
  const float* question = (const float*)d_in[0];
  const float* answer   = (const float*)d_in[1];
  const float* x        = (const float*)d_in[2];
  const float* h0       = (const float*)d_in[3];
  const float* c0       = (const float*)d_in[4];
  const float* W_qh     = (const float*)d_in[5];
  const float* W_ah     = (const float*)d_in[6];
  const float* W_ih     = (const float*)d_in[7];
  const float* W_hh     = (const float*)d_in[8];
  const float* b_qh     = (const float*)d_in[9];
  const float* b_ah     = (const float*)d_in[10];
  const float* b_ih     = (const float*)d_in[11];
  const float* b_hh     = (const float*)d_in[12];
  float* out = (float*)d_out;

  // ---- workspace carve (~740MB) ----
  char* ws = (char*)d_ws;
  size_t off = 0;
  auto alloc = [&](size_t bytes) -> char* {
    char* p = ws + off;
    off = (off + bytes + 255) & ~(size_t)255;
    return p;
  };
  const size_t WB = (size_t)G4 * HID;                       // elems per layer weight
  __bf16* WqhT = (__bf16*)alloc(LAYERS * WB * 2);
  __bf16* WahT = (__bf16*)alloc(LAYERS * WB * 2);
  __bf16* WihT = (__bf16*)alloc(LAYERS * WB * 2);
  __bf16* WhhT = (__bf16*)alloc(LAYERS * WB * 2);
  __bf16* q0bf = (__bf16*)alloc((size_t)BATCH * HID * 2);
  __bf16* a0bf = (__bf16*)alloc((size_t)BATCH * HID * 2);
  float*  bias = (float*)alloc((size_t)LAYERS * G4 * 4);
  float*  qa   = (float*)alloc((size_t)BATCH * G4 * 4);
  float*  xg   = (float*)alloc((size_t)T_STEPS * BATCH * G4 * 4);
  __bf16* act0 = (__bf16*)alloc((size_t)T_STEPS * BATCH * HID * 2);
  __bf16* act1 = (__bf16*)alloc((size_t)T_STEPS * BATCH * HID * 2);
  __bf16* hbf  = (__bf16*)alloc((size_t)2 * BATCH * HID * 2);
  float*  cst  = (float*)alloc((size_t)BATCH * HID * 4);
  unsigned* bar = (unsigned*)alloc(256);

  // ---- one-time converts / transposes (weights fit in 192MB L2 afterwards) ----
  k_convert_bf16<<<512, 256, 0, stream>>>(x, act0, (long)T_STEPS * BATCH * HID);
  k_convert_bf16<<<32, 256, 0, stream>>>(question, q0bf, (long)BATCH * HID); // question[0]
  k_convert_bf16<<<32, 256, 0, stream>>>(answer,   a0bf, (long)BATCH * HID); // answer[0]
  k_bias_sum<<<(LAYERS * G4 + 255) / 256, 256, 0, stream>>>(b_qh, b_ah, b_ih, b_hh,
                                                            bias, LAYERS * G4);
  dim3 tgrid(G4 / 64, HID / 64);
  for (int l = 0; l < LAYERS; ++l) {
    k_transpose_bf16<<<tgrid, 256, 0, stream>>>(W_qh + l * WB, WqhT + l * WB, HID, G4);
    k_transpose_bf16<<<tgrid, 256, 0, stream>>>(W_ah + l * WB, WahT + l * WB, HID, G4);
    k_transpose_bf16<<<tgrid, 256, 0, stream>>>(W_ih + l * WB, WihT + l * WB, HID, G4);
    k_transpose_bf16<<<tgrid, 256, 0, stream>>>(W_hh + l * WB, WhhT + l * WB, HID, G4);
  }

  const size_t smemA   = (size_t)64 * 1024 * 2;             // 128KB A stage
  const size_t smemRec = smemA + (size_t)64 * 128 * 4;      // +32KB gate exchange

  for (int l = 0; l < LAYERS; ++l) {
    k_gemm_qa<<<G4 / 128, 256, smemA, stream>>>(q0bf, a0bf, WqhT + l * WB,
                                                WahT + l * WB, bias + (size_t)l * G4, qa);
    const __bf16* in_act = (l == 0) ? act0 : act1;
    dim3 xgrid(G4 / 128, T_STEPS * BATCH / 64);
    k_gemm_xg<<<xgrid, 256, smemA, stream>>>(in_act, WihT + l * WB, qa, xg);

    k_init_state<<<(BATCH * HID + 255) / 256, 256, 0, stream>>>(
        h0 + (size_t)l * BATCH * HID, c0 + (size_t)l * BATCH * HID, hbf, cst, bar);

    float*  hs_f32 = (l == LAYERS - 1) ? out : nullptr;
    __bf16* hs_bf  = (l == 0) ? act1 : nullptr;
    float*  hlast  = out + (size_t)T_STEPS * BATCH * HID + (size_t)l * BATCH * HID;
    float*  clast  = out + (size_t)T_STEPS * BATCH * HID + (size_t)LAYERS * BATCH * HID
                         + (size_t)l * BATCH * HID;
    k_recurrent<<<NB_REC, 256, smemRec, stream>>>(xg, WhhT + l * WB, hbf, cst,
                                                  hs_bf, hs_f32, hlast, clast, bar);
  }
}
